// MultiHeadAttention_10402410791095
// MI455X (gfx1250) — matrix operand order repaired and checked by reading, not compiled, not run on hardware
//
#include <hip/hip_runtime.h>
#include <hip/hip_bf16.h>

// ---------------------------------------------------------------------------
// Types for CDNA5 WMMA (wave32)
// ---------------------------------------------------------------------------
typedef __bf16 bf16;
typedef bf16  v16bf __attribute__((ext_vector_type(16)));
typedef bf16  v8bf  __attribute__((ext_vector_type(8)));
typedef float v8f   __attribute__((ext_vector_type(8)));

#define WMMA_BF16(a, b, c) \
  __builtin_amdgcn_wmma_f32_16x16x32_bf16(false, (a), false, (b), (short)0, (c), false, false)

// Model constants
#define BATCH   2
#define SEQ     2048
#define DMODEL  2048
#define NHEADS  32
#define NKV     8
#define DHEAD   64
#define CHUNK   512

// ---------------------------------------------------------------------------
// Fragment loaders (bf16 16x16x32 WMMA, wave32)
//
// A (16x32, MxK): lane = row (lane&15); half 0 holds K 0-15 (elems 0-15),
//                 half 1 holds K 16-31.
// B (32x16, KxN): lane = col (lane&15); half 0 holds K 0-15 (elems 0-15),
//                 half 1 holds K 16-31.  Source stored as (N x K) row-major,
//                 so a column of B is a contiguous row of the source.
// ---------------------------------------------------------------------------
__device__ inline v16bf load_afrag(const bf16* base, int ld, int lane, int k) {
  int row  = lane & 15;
  int koff = (lane < 16) ? 0 : 16;
  const bf16* p = base + (size_t)row * ld + k + koff;
  union { v16bf v; v8bf h[2]; } u;
  u.h[0] = *(const v8bf*)(p);
  u.h[1] = *(const v8bf*)(p + 8);
  return u.v;
}

__device__ inline v16bf load_bfrag(const bf16* base, int ld, int lane, int k) {
  int col   = lane & 15;
  int kbase = (lane < 16) ? 0 : 16;
  const bf16* p = base + (size_t)col * ld + k + kbase;
  union { v16bf v; v8bf h[2]; } u;
  u.h[0] = *(const v8bf*)(p);
  u.h[1] = *(const v8bf*)(p + 8);
  return u.v;
}

// ---------------------------------------------------------------------------
// fp32 -> bf16 conversion (grid-stride)
// ---------------------------------------------------------------------------
__global__ void cvt_f32_bf16(const float* __restrict__ src, bf16* __restrict__ dst, int n) {
  int i = blockIdx.x * blockDim.x + threadIdx.x;
  int stride = gridDim.x * blockDim.x;
  for (; i < n; i += stride) dst[i] = (bf16)src[i];
}

// ---------------------------------------------------------------------------
// Block-tiled GEMM: C[m][n] = sum_k A[m][k] * B[n][k]
//   A: MxK bf16 row-major, B: NxK bf16 row-major (weight), C: MxN f32.
// Block = 256 threads (8 waves) computes a 128x128 C tile.
// Per k-step (BK=32): stage A-panel 128x32 and B-panel 128x32 into LDS
// (padded stride 40 elems = 80B -> fragment ds_load_b128 is bank-conflict
// free), each wave computes a 32(M)x64(N) sub-tile = 8 chained WMMAs.
// Next panel's global loads are issued before the WMMA block (prefetch).
// Requires M%128==0, N%128==0, K%32==0 (true for all calls here).
// ---------------------------------------------------------------------------
#define BM 128
#define BN 128
#define BK 32
#define LSTR 40   // LDS row stride in elements (32 + 8 pad)

__global__ void gemm_bf16_wmma(const bf16* __restrict__ A, const bf16* __restrict__ B,
                               float* __restrict__ C, int M, int N, int K) {
  __shared__ __attribute__((aligned(32))) bf16 As[BM * LSTR];
  __shared__ __attribute__((aligned(32))) bf16 Bs[BN * LSTR];

  int tid  = threadIdx.x;
  int lane = tid & 31;
  int w    = tid >> 5;

  int nblk = N / BN;
  int bm = blockIdx.x / nblk;
  int bn = blockIdx.x - bm * nblk;
  int m0 = bm * BM;
  int n0 = bn * BN;

  int wm = w & 3;    // wave's 32-row slab within the 128-row tile
  int wn = w >> 2;   // wave's 64-col slab within the 128-col tile

  // Cooperative staging: each thread owns one 16-element chunk of A and of B.
  int lrow = tid >> 1;          // 0..127
  int lcol = (tid & 1) * 16;    // 0 or 16

  const bf16* Ag = A + (size_t)(m0 + lrow) * K + lcol;
  const bf16* Bg = B + (size_t)(n0 + lrow) * K + lcol;

  // Prefetch panel k=0 into registers.
  v8bf ap0 = *(const v8bf*)(Ag);
  v8bf ap1 = *(const v8bf*)(Ag + 8);
  v8bf bp0 = *(const v8bf*)(Bg);
  v8bf bp1 = *(const v8bf*)(Bg + 8);

  v8f acc[2][4] = {};

  for (int kt = 0; kt < K; kt += BK) {
    __syncthreads();   // previous step's fragment reads complete
    *(v8bf*)(&As[lrow * LSTR + lcol])     = ap0;
    *(v8bf*)(&As[lrow * LSTR + lcol + 8]) = ap1;
    *(v8bf*)(&Bs[lrow * LSTR + lcol])     = bp0;
    *(v8bf*)(&Bs[lrow * LSTR + lcol + 8]) = bp1;
    __syncthreads();   // panel visible to all waves

    // Issue next panel's global loads now; they overlap the WMMAs below.
    if (kt + BK < K) {
      int ko = kt + BK;
      ap0 = *(const v8bf*)(Ag + ko);
      ap1 = *(const v8bf*)(Ag + ko + 8);
      bp0 = *(const v8bf*)(Bg + ko);
      bp1 = *(const v8bf*)(Bg + ko + 8);
    }

    v16bf af0 = load_afrag(&As[(wm * 32)      * LSTR], LSTR, lane, 0);
    v16bf af1 = load_afrag(&As[(wm * 32 + 16) * LSTR], LSTR, lane, 0);
#pragma unroll
    for (int t = 0; t < 4; ++t) {
      v16bf bb = load_bfrag(&Bs[(wn * 64 + t * 16) * LSTR], LSTR, lane, 0);
      acc[0][t] = WMMA_BF16(af0, bb, acc[0][t]);
      acc[1][t] = WMMA_BF16(af1, bb, acc[1][t]);
    }
  }

  int half = lane >> 4;
  int col  = lane & 15;
#pragma unroll
  for (int s = 0; s < 2; ++s)
#pragma unroll
    for (int t = 0; t < 4; ++t)
#pragma unroll
      for (int r = 0; r < 8; ++r)
        C[(size_t)(m0 + wm * 32 + s * 16 + r + half * 8) * N +
          n0 + wn * 64 + t * 16 + col] = acc[s][t][r];
}

// ---------------------------------------------------------------------------
// Per-head prep: RMSNorm (+ RoPE on rope layers) and layout change.
// One wave handles one (b, s, head): 64 values, 2 per lane; each lane owns
// exactly one RoPE pair (2*lane, 2*lane+1).
//   transposed == 0 : dst[(b*H + h)*S + s][d]         (Q, K: head-major rows)
//   transposed == 1 : dst[((b*H + h)*64 + d)*S + s]   (V: per-head (d, s))
// ---------------------------------------------------------------------------
__global__ void prep_heads(const float* __restrict__ raw, const float* __restrict__ gamma,
                           bf16* __restrict__ dst, const int* __restrict__ idx_p,
                           int H, int do_norm_rope, int transposed) {
  int lane = threadIdx.x & 31;
  int wid  = blockIdx.x * (blockDim.x >> 5) + (threadIdx.x >> 5);
  int h = wid % H;
  int s = (wid / H) % SEQ;
  int b = wid / (H * SEQ);
  int d0 = lane * 2;

  const float* p = raw + (((size_t)b * SEQ + s) * H + h) * DHEAD + d0;
  float x1 = p[0], x2 = p[1];

  if (do_norm_rope) {
    float ss = x1 * x1 + x2 * x2;
#pragma unroll
    for (int off = 16; off; off >>= 1) ss += __shfl_xor(ss, off, 32);
    float inv = rsqrtf(ss * (1.0f / (float)DHEAD) + 1e-6f);
    x1 = x1 * inv * gamma[d0];
    x2 = x2 * inv * gamma[d0 + 1];
    int idx = idx_p[0];
    if (((idx + 1) & 3) != 0) {            // ROPE_PERIOD == 4
      float theta = __powf(10000.0f, -(float)d0 * (1.0f / (float)DHEAD));
      float ang = (float)s * theta;
      float sn, c;
      __sincosf(ang, &sn, &c);
      float r1 = x1 * c - x2 * sn;
      float r2 = x1 * sn + x2 * c;
      x1 = r1; x2 = r2;
    }
  }
  if (!transposed) {
    bf16* q = dst + (((size_t)b * H + h) * SEQ + s) * DHEAD + d0;
    q[0] = (bf16)x1;
    q[1] = (bf16)x2;
  } else {
    dst[(((size_t)b * H + h) * DHEAD + d0)     * SEQ + s] = (bf16)x1;
    dst[(((size_t)b * H + h) * DHEAD + d0 + 1) * SEQ + s] = (bf16)x2;
  }
}

// ---------------------------------------------------------------------------
// Flash attention, one wave per (b, h, 16-query tile).
// Q: (b, h, s, 64) bf16; K: (b, kvh, s, 64) bf16; Vt: (b, kvh, 64, s) bf16.
// Rope layer: keys restricted to the 512-chunk of the query (causal inside).
// Non-rope layer: full causal + softplus temperature folded into scale.
// Scores via 2 chained WMMAs over d=64; P goes through a 1KB/wave LDS tile
// to convert C-fragment layout -> A-fragment layout; PV via 4 WMMAs / 32 keys.
// ---------------------------------------------------------------------------
__global__ void attn_wmma(const bf16* __restrict__ Q, const bf16* __restrict__ Kc,
                          const bf16* __restrict__ Vt, bf16* __restrict__ O,
                          const int* __restrict__ idx_p, const float* __restrict__ temp_p) {
  __shared__ __attribute__((aligned(32))) bf16 pbuf[8][16 * 32];

  int lane = threadIdx.x & 31;
  int w    = threadIdx.x >> 5;
  int wid  = blockIdx.x * 8 + w;

  const int QTILES = SEQ / 16;                 // 128
  int qt = wid % QTILES;
  int h  = (wid / QTILES) % NHEADS;
  int b  = wid / (QTILES * NHEADS);
  int m0 = qt * 16;
  int kvh = h >> 2;                            // n_rep = 4

  int idx = idx_p[0];
  bool rope = ((idx + 1) & 3) != 0;
  float sc = 0.125f;                           // 1/sqrt(64)
  if (!rope) {
    float t  = temp_p[0];
    float temp = logf(1.0f + __expf(t)) + 0.5f;  // softplus + 0.5
    sc /= temp;
  }
  int start = rope ? (m0 & ~(CHUNK - 1)) : 0;

  const bf16* qb = Q + (((size_t)b * NHEADS + h) * SEQ + m0) * DHEAD;
  v16bf qa0 = load_afrag(qb, DHEAD, lane, 0);
  v16bf qa1 = load_afrag(qb, DHEAD, lane, 32);

  const bf16* kbase = Kc + ((size_t)b * NKV + kvh) * SEQ * DHEAD;
  const bf16* vbase = Vt + ((size_t)b * NKV + kvh) * DHEAD * SEQ;

  float mi[8], li[8];
#pragma unroll
  for (int r = 0; r < 8; ++r) { mi[r] = -1e30f; li[r] = 0.0f; }
  v8f acc[4] = {};

  int half = lane >> 4;
  int col  = lane & 15;

  for (int kb = start; kb < m0 + 16; kb += 32) {
    // ---- scores: two 16x16 tiles (keys kb..kb+15, kb+16..kb+31) ----
    v8f s0 = {}, s1 = {};
    {
      v16bf b00 = load_bfrag(kbase + (size_t)kb * DHEAD, DHEAD, lane, 0);
      v16bf b01 = load_bfrag(kbase + (size_t)kb * DHEAD, DHEAD, lane, 32);
      s0 = WMMA_BF16(qa0, b00, s0);
      s0 = WMMA_BF16(qa1, b01, s0);
      v16bf b10 = load_bfrag(kbase + (size_t)(kb + 16) * DHEAD, DHEAD, lane, 0);
      v16bf b11 = load_bfrag(kbase + (size_t)(kb + 16) * DHEAD, DHEAD, lane, 32);
      s1 = WMMA_BF16(qa0, b10, s1);
      s1 = WMMA_BF16(qa1, b11, s1);
    }

    int c0g = kb + col;
    int c1g = kb + 16 + col;
#pragma unroll
    for (int r = 0; r < 8; ++r) {
      int row = m0 + r + half * 8;
      float v0 = (c0g <= row) ? s0[r] * sc : -1e30f;
      float v1 = (c1g <= row) ? s1[r] * sc : -1e30f;

      // row max across the 16 lanes of this half (xor masks stay in-half)
      float bm = fmaxf(v0, v1);
#pragma unroll
      for (int off = 8; off; off >>= 1) bm = fmaxf(bm, __shfl_xor(bm, off, 32));
      float mnew  = fmaxf(mi[r], bm);
      float alpha = __expf(mi[r] - mnew);
      float p0 = __expf(v0 - mnew);
      float p1 = __expf(v1 - mnew);
      float rs = p0 + p1;
#pragma unroll
      for (int off = 8; off; off >>= 1) rs += __shfl_xor(rs, off, 32);
      li[r] = li[r] * alpha + rs;
      mi[r] = mnew;
#pragma unroll
      for (int t = 0; t < 4; ++t) acc[t][r] *= alpha;

      int lr = r + half * 8;
      pbuf[w][lr * 32 + col]      = (bf16)p0;
      pbuf[w][lr * 32 + 16 + col] = (bf16)p1;
    }

    // ---- P (16x32) as A-fragment from LDS; PV over 4 d-tiles ----
    v16bf pa = load_afrag(&pbuf[w][0], 32, lane, 0);
#pragma unroll
    for (int t = 0; t < 4; ++t) {
      v16bf vb = load_bfrag(vbase + (size_t)(t * 16) * SEQ, SEQ, lane, kb);
      acc[t] = WMMA_BF16(pa, vb, acc[t]);
    }
  }

  // ---- epilogue: divide by row sums, store bf16 (b, s, h*64+d) ----
  bf16* ob = O + (((size_t)b * SEQ + m0) * DMODEL) + h * DHEAD;
#pragma unroll
  for (int t = 0; t < 4; ++t)
#pragma unroll
    for (int r = 0; r < 8; ++r) {
      float o = acc[t][r] / li[r];
      ob[(size_t)(r + half * 8) * DMODEL + t * 16 + col] = (bf16)o;
    }
}

// ---------------------------------------------------------------------------
// Host orchestration
// ---------------------------------------------------------------------------
extern "C" void kernel_launch(void* const* d_in, const int* in_sizes, int n_in,
                              void* d_out, int out_size, void* d_ws, size_t ws_size,
                              hipStream_t stream) {
  (void)in_sizes; (void)n_in; (void)out_size; (void)ws_size;

  const float* x   = (const float*)d_in[0];
  const int*   idx = (const int*)  d_in[1];
  const float* Wq  = (const float*)d_in[2];
  const float* Wk  = (const float*)d_in[3];
  const float* Wv  = (const float*)d_in[4];
  const float* Wo  = (const float*)d_in[5];
  const float* gq  = (const float*)d_in[6];
  const float* gk  = (const float*)d_in[7];
  const float* ts  = (const float*)d_in[8];
  float* out = (float*)d_out;

  const size_t NX   = (size_t)BATCH * SEQ * DMODEL;        // 8.4M
  const size_t NWQ  = (size_t)DMODEL * DMODEL;             // 4.2M
  const size_t NWK  = (size_t)(NKV * DHEAD) * DMODEL;      // 1.05M
  const size_t NQH  = (size_t)BATCH * NHEADS * SEQ * DHEAD;
  const size_t NKH  = (size_t)BATCH * NKV * SEQ * DHEAD;

  char* ws = (char*)d_ws;
  size_t off = 0;
  auto carve = [&](size_t bytes) -> void* {
    void* p = ws + off;
    off += (bytes + 255) & ~(size_t)255;
    return p;
  };
  bf16*  xb   = (bf16*) carve(NX  * 2);
  bf16*  wqb  = (bf16*) carve(NWQ * 2);
  bf16*  wkb  = (bf16*) carve(NWK * 2);
  bf16*  wvb  = (bf16*) carve(NWK * 2);
  bf16*  wob  = (bf16*) carve(NWQ * 2);
  float* qraw = (float*)carve(NX  * 4);
  float* kraw = (float*)carve(NKH * 4);
  float* vraw = (float*)carve(NKH * 4);
  bf16*  qh   = (bf16*) carve(NQH * 2);
  bf16*  kh   = (bf16*) carve(NKH * 2);
  bf16*  vt   = (bf16*) carve(NKH * 2);
  bf16*  ao   = (bf16*) carve(NX  * 2);

  // 1) fp32 -> bf16 conversions
  cvt_f32_bf16<<<2048, 256, 0, stream>>>(x,  xb,  (int)NX);
  cvt_f32_bf16<<<1024, 256, 0, stream>>>(Wq, wqb, (int)NWQ);
  cvt_f32_bf16<<< 512, 256, 0, stream>>>(Wk, wkb, (int)NWK);
  cvt_f32_bf16<<< 512, 256, 0, stream>>>(Wv, wvb, (int)NWK);
  cvt_f32_bf16<<<1024, 256, 0, stream>>>(Wo, wob, (int)NWQ);

  // 2) QKV projections (M = B*S = 4096); 128x128 C tile per block
  const int M = BATCH * SEQ;
  {
    int blocks_q  = (M / BM) * (DMODEL / BN);               // 32*16 = 512
    gemm_bf16_wmma<<<blocks_q, 256, 0, stream>>>(xb, wqb, qraw, M, DMODEL, DMODEL);
    int blocks_kv = (M / BM) * ((NKV * DHEAD) / BN);        // 32*4  = 128
    gemm_bf16_wmma<<<blocks_kv, 256, 0, stream>>>(xb, wkb, kraw, M, NKV * DHEAD, DMODEL);
    gemm_bf16_wmma<<<blocks_kv, 256, 0, stream>>>(xb, wvb, vraw, M, NKV * DHEAD, DMODEL);
  }

  // 3) RMSNorm (+RoPE) and layout: one wave per (b, s, head)
  {
    int q_blocks = (BATCH * SEQ * NHEADS) / 8;  // 16384
    int k_blocks = (BATCH * SEQ * NKV)    / 8;  // 4096
    prep_heads<<<q_blocks, 256, 0, stream>>>(qraw, gq, qh, idx, NHEADS, 1, 0);
    prep_heads<<<k_blocks, 256, 0, stream>>>(kraw, gk, kh, idx, NKV,    1, 0);
    prep_heads<<<k_blocks, 256, 0, stream>>>(vraw, gk, vt, idx, NKV,    0, 1);
  }

  // 4) Flash attention: one wave per (b, h, 16-query tile)
  {
    int blocks = (BATCH * NHEADS * (SEQ / 16)) / 8;         // 1024
    attn_wmma<<<blocks, 256, 0, stream>>>(qh, kh, vt, ao, idx, ts);
  }

  // 5) Output projection -> d_out (f32)
  {
    int blocks = (M / BM) * (DMODEL / BN);                  // 512
    gemm_bf16_wmma<<<blocks, 256, 0, stream>>>(ao, wob, out, M, DMODEL, DMODEL);
  }
}
